// HistoryAwarePointerDecoder_30906584662765
// MI455X (gfx1250) — matrix operand order repaired and checked by reading, not compiled, not run on hardware
//
#include <hip/hip_runtime.h>
#include <hip/hip_bf16.h>
#include <stdint.h>

typedef __attribute__((ext_vector_type(16))) __bf16 v16bf;
typedef __attribute__((ext_vector_type(8)))  float  v8f;

#define WGSZ   256
#define TM     64
#define TN     128
#define TK     32
#define APITCH 56      // bf16 elems per A-tile row (112B, 16B-aligned rows)
#define BPITCH 56
#define CPITCH 132     // f32 elems per C-tile row (528B = 16*33)
#define SMEMB  33792   // max( A(7168)+B(14336)=21504 , C 64*132*4=33792 )

// round-to-nearest-even f32 -> bf16, packed pair (lo = first)
__device__ __forceinline__ uint32_t f2bf_pk(float a, float b) {
  uint32_t ua = __float_as_uint(a);
  uint32_t ub = __float_as_uint(b);
  ua = (ua + 0x7FFFu + ((ua >> 16) & 1u)) >> 16;
  ub = (ub + 0x7FFFu + ((ub >> 16) & 1u)) >> 16;
  return (ua & 0xFFFFu) | ((ub & 0xFFFFu) << 16);
}

union Frag { uint4 u[2]; v16bf v; };

// C[M,N] = A[M,K] @ W[N,K]^T + bias  (bf16 WMMA, f32 accum)
// mode 0: plain store (optionally += existing C)     [any N multiple of 128]
// mode 1: fused LayerNorm(gamma,beta) (+ReLU)        [requires N == 128]
// mode 2: fused LN + score = tanh(query+k)@wv + wvb, masked, C treated as [B*S]
__global__ __launch_bounds__(WGSZ)
void gemm_kernel(const float* __restrict__ A, int ldA,
                 const float* __restrict__ W, int ldW,
                 const float* __restrict__ bias,
                 float* __restrict__ C, int ldc,
                 int K, int mode, int relu, int accum,
                 const float* __restrict__ gamma, const float* __restrict__ beta,
                 const float* __restrict__ query, const float* __restrict__ wv,
                 const float* __restrict__ wvb,
                 const int* __restrict__ mask, const int* __restrict__ allmask,
                 int S) {
  __shared__ __align__(16) unsigned char smem[SMEMB];
  unsigned short* As = (unsigned short*)smem;
  unsigned short* Bs = As + TM * APITCH;       // byte 7168, 16B aligned
  float*          Cf = (float*)smem;

  const int tid  = threadIdx.x;
  const int lane = tid & 31;
  const int wave = tid >> 5;
  const int m0 = blockIdx.y * TM;
  const int n0 = blockIdx.x * TN;
  const int wm = (wave >> 2) * 32;             // 2 wave-rows
  const int wn = (wave & 3) * 32;              // 4 wave-cols

  const v8f vzero = {0.f,0.f,0.f,0.f,0.f,0.f,0.f,0.f};
  v8f acc[2][2];
  acc[0][0]=vzero; acc[0][1]=vzero; acc[1][0]=vzero; acc[1][1]=vzero;

  const int arow = tid >> 2, acol = (tid & 3) * 8;   // A stage: 64x32
  const int brow = tid >> 1, bcol = (tid & 1) * 16;  // B stage: 128x32
  const int am = lane & 15, ah = lane >> 4;

  for (int k0 = 0; k0 < K; k0 += TK) {
    __syncthreads();
    { // stage A tile (f32 -> bf16)
      const float* src = A + (size_t)(m0 + arow) * ldA + k0 + acol;
      float4 f0 = *(const float4*)(src);
      float4 f1 = *(const float4*)(src + 4);
      uint4 p;
      p.x = f2bf_pk(f0.x, f0.y); p.y = f2bf_pk(f0.z, f0.w);
      p.z = f2bf_pk(f1.x, f1.y); p.w = f2bf_pk(f1.z, f1.w);
      *(uint4*)(As + arow * APITCH + acol) = p;
    }
    { // stage B tile (weights row-major [N][K])
      const float* src = W + (size_t)(n0 + brow) * ldW + k0 + bcol;
      float4 f0 = *(const float4*)(src);
      float4 f1 = *(const float4*)(src + 4);
      float4 f2 = *(const float4*)(src + 8);
      float4 f3 = *(const float4*)(src + 12);
      uint4 p0, p1;
      p0.x = f2bf_pk(f0.x, f0.y); p0.y = f2bf_pk(f0.z, f0.w);
      p0.z = f2bf_pk(f1.x, f1.y); p0.w = f2bf_pk(f1.z, f1.w);
      p1.x = f2bf_pk(f2.x, f2.y); p1.y = f2bf_pk(f2.z, f2.w);
      p1.z = f2bf_pk(f3.x, f3.y); p1.w = f2bf_pk(f3.z, f3.w);
      *(uint4*)(Bs + brow * BPITCH + bcol)     = p0;
      *(uint4*)(Bs + brow * BPITCH + bcol + 8) = p1;
    }
    __syncthreads();

    Frag a[2], b[2];
#pragma unroll
    for (int i = 0; i < 2; i++) {  // A 16x32: lanes halves split K 0-15/16-31
      const unsigned short* p = As + (wm + i*16 + am) * APITCH + ah * 16;
      a[i].u[0] = *(const uint4*)p;
      a[i].u[1] = *(const uint4*)(p + 8);
    }
#pragma unroll
    for (int j = 0; j < 2; j++) {  // B 32x16: lanes halves split K 0-15/16-31
      const unsigned short* p = Bs + (wn + j*16 + am) * BPITCH + ah * 16;
      b[j].u[0] = *(const uint4*)p;
      b[j].u[1] = *(const uint4*)(p + 8);
    }
#pragma unroll
    for (int i = 0; i < 2; i++)
#pragma unroll
      for (int j = 0; j < 2; j++)
        acc[i][j] = __builtin_amdgcn_wmma_f32_16x16x32_bf16(
            false, a[i].v, false, b[j].v, (short)0, acc[i][j], false, false);
  }
  __syncthreads();

  if (mode == 0) {
#pragma unroll
    for (int i = 0; i < 2; i++)
#pragma unroll
      for (int j = 0; j < 2; j++)
#pragma unroll
        for (int v = 0; v < 8; v++) {
          int gr = m0 + wm + i*16 + (lane >> 4) * 8 + v;
          int gc = n0 + wn + j*16 + (lane & 15);
          float val = acc[i][j][v] + bias[gc];
          if (accum) val += C[(size_t)gr * ldc + gc];
          C[(size_t)gr * ldc + gc] = val;
        }
    return;
  }

  // modes 1/2: dump (acc + bias) into LDS, LayerNorm each 128-wide row
#pragma unroll
  for (int i = 0; i < 2; i++)
#pragma unroll
    for (int j = 0; j < 2; j++)
#pragma unroll
      for (int v = 0; v < 8; v++) {
        int lr = wm + i*16 + (lane >> 4) * 8 + v;
        int lc = wn + j*16 + (lane & 15);
        Cf[lr * CPITCH + lc] = acc[i][j][v] + bias[n0 + lc];
      }
  __syncthreads();

  const int lrow = tid >> 2;       // 64 rows, 4 threads each (32 cols/thread)
  const int q4   = tid & 3;
  float x[32], s = 0.f, ss = 0.f;
#pragma unroll
  for (int u = 0; u < 8; u++) {
    float4 f = *(float4*)&Cf[lrow * CPITCH + q4 * 32 + u * 4];
    x[u*4+0] = f.x; x[u*4+1] = f.y; x[u*4+2] = f.z; x[u*4+3] = f.w;
    s  += f.x + f.y + f.z + f.w;
    ss += f.x*f.x + f.y*f.y + f.z*f.z + f.w*f.w;
  }
  s  += __shfl_xor(s, 1);  s  += __shfl_xor(s, 2);
  ss += __shfl_xor(ss, 1); ss += __shfl_xor(ss, 2);
  float mean = s * (1.0f / 128.0f);
  float var  = ss * (1.0f / 128.0f) - mean * mean;
  float rstd = rsqrtf(var + 1e-5f);
  int gr = m0 + lrow;

  if (mode == 1) {
#pragma unroll
    for (int u = 0; u < 32; u++) {
      int c = q4 * 32 + u;
      float y = (x[u] - mean) * rstd * gamma[c] + beta[c];
      if (relu) y = fmaxf(y, 0.0f);
      C[(size_t)gr * ldc + c] = y;
    }
  } else { // mode 2: pointer score
    int b = gr / S, si = gr - b * S;
    const float* qrow = query + (size_t)b * 128;
    float p = 0.f;
#pragma unroll
    for (int u = 0; u < 32; u++) {
      int c = q4 * 32 + u;
      float y = (x[u] - mean) * rstd * gamma[c] + beta[c];
      p += tanhf(qrow[c] + y) * wv[c];
    }
    p += __shfl_xor(p, 1); p += __shfl_xor(p, 2);
    if (q4 == 0) {
      float sc = p + wvb[0];
      if (allmask[b])         sc = (si == 0) ? 0.0f : -1e9f;
      else if (mask[gr] == 0) sc = -1e9f;
      C[gr] = sc;
    }
  }
}

// ---------- small helper kernels ----------

__global__ void build_hist_kernel(const float* __restrict__ hemb,
                                  const float* __restrict__ enc,
                                  const int*   __restrict__ last,
                                  float* __restrict__ hist,
                                  float* __restrict__ cur) {
  int idx = blockIdx.x * blockDim.x + threadIdx.x;   // (B*51)*64 float4s
  int row = idx >> 6, q = idx & 63;
  if (row >= 512 * 51) return;
  int b = row / 51, t = row - b * 51;
  float4 v;
  if (t < 50) v = *(const float4*)(hemb + ((size_t)(b * 50 + t)) * 256 + q * 4);
  else {
    v = *(const float4*)(enc + ((size_t)b * 200 + last[b]) * 256 + q * 4);
    *(float4*)(cur + (size_t)b * 256 + q * 4) = v;
  }
  *(float4*)(hist + (size_t)row * 256 + q * 4) = v;
}

__global__ void allmask_kernel(const int* __restrict__ mask, int* __restrict__ am) {
  int b = blockIdx.x * blockDim.x + threadIdx.x;
  if (b >= 512) return;
  int s = 0;
  for (int i = 0; i < 200; i++) s += mask[b * 200 + i];
  am[b] = (s == 0) ? 1 : 0;
}

__global__ __launch_bounds__(64)
void attn_kernel(const float* __restrict__ qkv, float* __restrict__ ctx) {
  const int b = blockIdx.x, h = blockIdx.y;
  __shared__ float Qs[51 * 33], Ks[51 * 33], Vs[51 * 33];
  const int t = threadIdx.x;
  if (t < 51) {
    const float* base = qkv + ((size_t)(b * 51) + t) * 384;
#pragma unroll
    for (int d = 0; d < 32; d++) {
      Qs[t * 33 + d] = base[h * 32 + d];
      Ks[t * 33 + d] = base[128 + h * 32 + d];
      Vs[t * 33 + d] = base[256 + h * 32 + d];
    }
  }
  __syncthreads();
  if (t >= 51) return;
  float qr[32];
#pragma unroll
  for (int d = 0; d < 32; d++) qr[d] = Qs[t * 33 + d] * 0.17677669529663689f;
  float sc[51], mx = -1e30f;
#pragma unroll
  for (int j = 0; j < 51; j++) {
    float dot = 0.f;
#pragma unroll
    for (int d = 0; d < 32; d++) dot += qr[d] * Ks[j * 33 + d];
    sc[j] = dot; mx = fmaxf(mx, dot);
  }
  float den = 0.f;
#pragma unroll
  for (int j = 0; j < 51; j++) { sc[j] = expf(sc[j] - mx); den += sc[j]; }
  float inv = 1.0f / den;
  float c[32];
#pragma unroll
  for (int d = 0; d < 32; d++) c[d] = 0.f;
#pragma unroll
  for (int j = 0; j < 51; j++) {
    float e = sc[j] * inv;
#pragma unroll
    for (int d = 0; d < 32; d++) c[d] += e * Vs[j * 33 + d];
  }
  float* out = ctx + ((size_t)(b * 51) + t) * 128 + h * 32;
#pragma unroll
  for (int d = 0; d < 32; d++) out[d] = c[d];
}

__global__ void lstm_gate_kernel(const float* __restrict__ g,
                                 const float* __restrict__ cprev,
                                 float* __restrict__ hout) {
  int b = blockIdx.x, j = threadIdx.x;
  const float* gr = g + (size_t)b * 512;
  float ii = 1.f / (1.f + expf(-gr[j]));
  float ff = 1.f / (1.f + expf(-gr[128 + j]));
  float gg = tanhf(gr[256 + j]);
  float oo = 1.f / (1.f + expf(-gr[384 + j]));
  float c  = ff * cprev[(size_t)b * 128 + j] + ii * gg;
  hout[(size_t)b * 128 + j] = oo * tanhf(c);
}

__global__ void summary_kernel(const float* __restrict__ ctx2,
                               float* __restrict__ summ) {
  int b = blockIdx.x, n = threadIdx.x;   // 128 threads
  float den = 0.f;
  for (int t = 0; t < 51; t++) den += expf((t + 1) * 0.5f - 25.5f);
  float acc = 0.f;
  for (int t = 0; t < 51; t++)
    acc += ctx2[((size_t)(b * 51) + t) * 128 + n] * (expf((t + 1) * 0.5f - 25.5f) / den);
  summ[(size_t)b * 128 + n] = acc;
}

__global__ void concat_kernel(const float* __restrict__ qb,
                              const float* __restrict__ summ,
                              float* __restrict__ cat) {
  int b = blockIdx.x, c = threadIdx.x;   // 256 threads
  cat[(size_t)b * 256 + c] =
      (c < 128) ? qb[(size_t)b * 128 + c] : summ[(size_t)b * 128 + (c - 128)];
}

// ---------- launch ----------

extern "C" void kernel_launch(void* const* d_in, const int* in_sizes, int n_in,
                              void* d_out, int out_size, void* d_ws, size_t ws_size,
                              hipStream_t stream) {
  const float* enc   = (const float*)d_in[0];
  const int*   mask  = (const int*)  d_in[1];
  const int*   last  = (const int*)  d_in[2];
  const float* h0    = (const float*)d_in[3];
  const float* c0    = (const float*)d_in[4];
  const float* hemb  = (const float*)d_in[5];
  const float* wih0  = (const float*)d_in[6];
  const float* whh0  = (const float*)d_in[7];
  const float* bih0  = (const float*)d_in[8];
  const float* bhh0  = (const float*)d_in[9];
  const float* wih1  = (const float*)d_in[10];
  const float* whh1  = (const float*)d_in[11];
  const float* bih1  = (const float*)d_in[12];
  const float* bhh1  = (const float*)d_in[13];
  const float* he_w  = (const float*)d_in[14];
  const float* he_b  = (const float*)d_in[15];
  const float* he_g  = (const float*)d_in[16];
  const float* he_be = (const float*)d_in[17];
  const float* miw   = (const float*)d_in[18];
  const float* mib   = (const float*)d_in[19];
  const float* mow   = (const float*)d_in[20];
  const float* mob   = (const float*)d_in[21];
  const float* agw   = (const float*)d_in[22];
  const float* agb   = (const float*)d_in[23];
  const float* agg_g = (const float*)d_in[24];
  const float* agg_be= (const float*)d_in[25];
  const float* wq_w  = (const float*)d_in[26];
  const float* wq_b  = (const float*)d_in[27];
  const float* wk_w  = (const float*)d_in[28];
  const float* wk_b  = (const float*)d_in[29];
  const float* wv_w  = (const float*)d_in[30];
  const float* wv_b  = (const float*)d_in[31];
  const float* lnq_g = (const float*)d_in[32];
  const float* lnq_b = (const float*)d_in[33];
  const float* lnk_g = (const float*)d_in[34];
  const float* lnk_b = (const float*)d_in[35];

  float* ws    = (float*)d_ws;
  float* hist  = ws + 0;          // 26112*256   (region reused by ctx/ctx2)
  float* ctx   = ws + 0;          // 26112*128
  float* ctx2  = ws + 3342336;    // 26112*128
  float* hc    = ws + 6684672;    // 26112*128
  float* qkv   = ws + 10027008;   // 26112*384
  float* cur   = ws + 20054016;   // 512*256
  float* summ  = ws + 20185088;   // 512*128
  float* g0    = ws + 20250624;   // 512*512
  float* h1    = ws + 20512768;   // 512*128
  float* g1    = ws + 20578304;   // 512*512
  float* qb    = ws + 20840448;   // 512*128
  float* cat   = ws + 20905984;   // 512*256
  float* aggo  = ws + 21037056;   // 512*128
  float* query = ws + 21102592;   // 512*128
  int*   amask = (int*)(ws + 21168128); // 512
  float* out   = (float*)d_out;

  build_hist_kernel<<<6528, 256, 0, stream>>>(hemb, enc, last, hist, cur);
  allmask_kernel<<<2, 256, 0, stream>>>(mask, amask);

  // hc = ReLU(LN(hist @ he_w^T + he_b))        [26112 x 128]
  gemm_kernel<<<dim3(1, 408), WGSZ, 0, stream>>>(hist, 256, he_w, 256, he_b, hc, 128,
      256, 1, 1, 0, he_g, he_be, nullptr, nullptr, nullptr, nullptr, nullptr, 0);
  // qkv = hc @ mha_in_w^T + b                  [26112 x 384]
  gemm_kernel<<<dim3(3, 408), WGSZ, 0, stream>>>(hc, 128, miw, 128, mib, qkv, 384,
      128, 0, 0, 0, nullptr, nullptr, nullptr, nullptr, nullptr, nullptr, nullptr, 0);
  attn_kernel<<<dim3(512, 4), 64, 0, stream>>>(qkv, ctx);
  // ctx2 = ctx @ mha_out_w^T + b               [26112 x 128]
  gemm_kernel<<<dim3(1, 408), WGSZ, 0, stream>>>(ctx, 128, mow, 128, mob, ctx2, 128,
      128, 0, 0, 0, nullptr, nullptr, nullptr, nullptr, nullptr, nullptr, nullptr, 0);
  summary_kernel<<<512, 128, 0, stream>>>(ctx2, summ);

  // LSTM layer 0
  gemm_kernel<<<dim3(4, 8), WGSZ, 0, stream>>>(cur, 256, wih0, 256, bih0, g0, 512,
      256, 0, 0, 0, nullptr, nullptr, nullptr, nullptr, nullptr, nullptr, nullptr, 0);
  gemm_kernel<<<dim3(4, 8), WGSZ, 0, stream>>>(h0, 128, whh0, 128, bhh0, g0, 512,
      128, 0, 0, 1, nullptr, nullptr, nullptr, nullptr, nullptr, nullptr, nullptr, 0);
  lstm_gate_kernel<<<512, 128, 0, stream>>>(g0, c0, h1);
  // LSTM layer 1
  gemm_kernel<<<dim3(4, 8), WGSZ, 0, stream>>>(h1, 128, wih1, 128, bih1, g1, 512,
      128, 0, 0, 0, nullptr, nullptr, nullptr, nullptr, nullptr, nullptr, nullptr, 0);
  gemm_kernel<<<dim3(4, 8), WGSZ, 0, stream>>>(h0 + 512 * 128, 128, whh1, 128, bhh1, g1, 512,
      128, 0, 0, 1, nullptr, nullptr, nullptr, nullptr, nullptr, nullptr, nullptr, 0);
  lstm_gate_kernel<<<512, 128, 0, stream>>>(g1, c0 + 512 * 128, qb);

  concat_kernel<<<512, 256, 0, stream>>>(qb, summ, cat);
  // agg = ReLU(LN(cat @ agg_w^T + agg_b))
  gemm_kernel<<<dim3(1, 8), WGSZ, 0, stream>>>(cat, 256, agw, 256, agb, aggo, 128,
      256, 1, 1, 0, agg_g, agg_be, nullptr, nullptr, nullptr, nullptr, nullptr, 0);
  // query = LN(agg @ wq_w^T + wq_b)
  gemm_kernel<<<dim3(1, 8), WGSZ, 0, stream>>>(aggo, 128, wq_w, 128, wq_b, query, 128,
      128, 1, 0, 0, lnq_g, lnq_b, nullptr, nullptr, nullptr, nullptr, nullptr, 0);

  // scores: keys GEMM with fused LN + tanh pointer score + masking -> d_out
  gemm_kernel<<<dim3(1, 1600), WGSZ, 0, stream>>>(enc, 256, wk_w, 256, wk_b, out, 1,
      256, 2, 0, 0, lnk_g, lnk_b, query, wv_w, wv_b, mask, amask, 200);
}
